// Graph_ConvLSTMCell_16741782520218
// MI455X (gfx1250) — matrix operand order repaired and checked by reading, not compiled, not run on hardware
//
#include <hip/hip_runtime.h>
#include <hip/hip_bf16.h>
#include <cstdint>

typedef _Float16 h8  __attribute__((ext_vector_type(8)));
typedef _Float16 v16h __attribute__((ext_vector_type(16)));
typedef float    v8f  __attribute__((ext_vector_type(8)));

#define NNODE 4096
#define BATCH 8

// ---------------------------------------------------------------------------
// CDNA5 async global->LDS copy (ASYNCcnt-tracked).  VDST VGPR holds the LDS
// byte address; low 32 bits of a generic shared pointer are the LDS address.
// ---------------------------------------------------------------------------
__device__ __forceinline__ void async_b128_to_lds(const _Float16* lds_dst,
                                                  const _Float16* gsrc) {
  const uint32_t l = (uint32_t)(uintptr_t)lds_dst;
  asm volatile("global_load_async_to_lds_b128 %0, %1, off"
               :
               : "v"(l), "v"((unsigned long long)(uintptr_t)gsrc)
               : "memory");
}
__device__ __forceinline__ void wait_async0() {
  asm volatile("s_wait_asynccnt 0" ::: "memory");
}

// ---------------------------------------------------------------------------
// WMMA helpers (CDNA5 v_wmma_f32_16x16x32_f16, wave32)
// ---------------------------------------------------------------------------
__device__ __forceinline__ v8f wmma16(v16h a, v16h b, v8f c) {
  return __builtin_amdgcn_wmma_f32_16x16x32_f16(false, a, false, b, (short)0, c,
                                                false, false);
}

// A fragment (16x32 f16): lane l -> row = base+(l&15); elements 0..7 hold
// K = kb..kb+7, elements 8..15 hold K = 16+kb..16+kb+7, kb = 8*(l>>4).
__device__ __forceinline__ v16h load_fragA(const _Float16* tile, int rowbase,
                                           int lane, int ld) {
  const int row = rowbase + (lane & 15);
  const int kb  = (lane >> 4) * 8;
  const _Float16* p = tile + row * ld + kb;
  h8 lo = *(const h8*)(p);
  h8 hi = *(const h8*)(p + 16);
  v16h r;
#pragma unroll
  for (int i = 0; i < 8; ++i) { r[i] = lo[i]; r[i + 8] = hi[i]; }
  return r;
}

// B fragment (32x16 f16): lane l -> col = base+(l&15); elements 0..7 hold
// K = kb..kb+7, elements 8..15 hold K = 16+kb..16+kb+7, kb = 8*(l>>4). Tile stored column-major [col][k].
__device__ __forceinline__ v16h load_fragB(const _Float16* tile, int colbase,
                                           int lane, int ld) {
  const int col = colbase + (lane & 15);
  const int kb  = (lane >> 4) * 8;
  const _Float16* p = tile + col * ld + kb;
  h8 lo = *(const h8*)(p);
  h8 hi = *(const h8*)(p + 16);
  v16h r;
#pragma unroll
  for (int i = 0; i < 8; ++i) { r[i] = lo[i]; r[i + 8] = hi[i]; }
  return r;
}

// ---------------------------------------------------------------------------
// K1: S1t[c][n] = sum_k concat(x,h)[b,n,k] * W1[k, c&63],  c = b*64+d
//     M = 32768 (b*4096+n), N = 64, K = 96 (3 steps; src selected per step)
// ---------------------------------------------------------------------------
__global__ __launch_bounds__(256) void gemm_in_kernel(
    const float* __restrict__ x, const float* __restrict__ h,
    const float* __restrict__ W1, _Float16* __restrict__ S1t) {
  constexpr int BM = 128, LD = 40;
  __shared__ _Float16 As[BM * LD];
  __shared__ _Float16 Bs[64 * LD];
  const int tid = threadIdx.x, lane = tid & 31, wave = tid >> 5;
  const int wm = wave & 3, wn = wave >> 2;  // 4 x 2 waves -> 32 rows x 32 cols
  const int m0 = blockIdx.x * BM;

  v8f acc[2][2];
#pragma unroll
  for (int i = 0; i < 2; ++i)
#pragma unroll
    for (int j = 0; j < 2; ++j) acc[i][j] = (v8f)0.f;

#pragma unroll
  for (int ks = 0; ks < 3; ++ks) {
    const float* src = (ks == 0) ? x : h;
    const int rs  = (ks == 0) ? 32 : 64;
    const int off = (ks == 0) ? 0 : (ks - 1) * 32;
    // A tile: 128 x 32 floats -> f16
    {
      const int c4 = (tid & 7) * 4;
      const int r  = tid >> 3;
#pragma unroll
      for (int p = 0; p < 4; ++p) {
        const int row = r + p * 32;
        const float4 a4 =
            *(const float4*)(src + (size_t)(m0 + row) * rs + off + c4);
        _Float16* d = &As[row * LD + c4];
        d[0] = (_Float16)a4.x; d[1] = (_Float16)a4.y;
        d[2] = (_Float16)a4.z; d[3] = (_Float16)a4.w;
      }
    }
    // B tile: Bs[col][k] = W1[(ks*32+k)*64 + col]
    {
      const int col = tid & 63;
      const int kb  = (tid >> 6) * 8;
#pragma unroll
      for (int i = 0; i < 8; ++i)
        Bs[col * LD + kb + i] = (_Float16)W1[(ks * 32 + kb + i) * 64 + col];
    }
    __syncthreads();
    v16h af[2], bf[2];
#pragma unroll
    for (int mi = 0; mi < 2; ++mi)
      af[mi] = load_fragA(As, wm * 32 + mi * 16, lane, LD);
#pragma unroll
    for (int ni = 0; ni < 2; ++ni)
      bf[ni] = load_fragB(Bs, wn * 32 + ni * 16, lane, LD);
#pragma unroll
    for (int mi = 0; mi < 2; ++mi)
#pragma unroll
      for (int ni = 0; ni < 2; ++ni)
        acc[mi][ni] = wmma16(af[mi], bf[ni], acc[mi][ni]);
    __syncthreads();
  }
  // Epilogue: write f16 transposed S1t[b*64+d][n], 8 contiguous n per lane.
#pragma unroll
  for (int mi = 0; mi < 2; ++mi) {
#pragma unroll
    for (int ni = 0; ni < 2; ++ni) {
      const int d     = wn * 32 + ni * 16 + (lane & 15);
      const int mbase = m0 + wm * 32 + mi * 16 + ((lane >> 4) * 8);
      const int b     = mbase >> 12;
      const int n     = mbase & (NNODE - 1);
      h8 o;
#pragma unroll
      for (int r = 0; r < 8; ++r) o[r] = (_Float16)acc[mi][ni][r];
      *(h8*)(S1t + (size_t)(b * 64 + d) * NNODE + n) = o;
    }
  }
}

// ---------------------------------------------------------------------------
// K2/K4: out[b][m][ch] = sum_n (adj[m][n]*mask[m][n]) * Bt[b*NCH+ch][n] + bias
//        M = 4096, N = BATCH*NCH, K = 4096. Bt is f16 [Ncols][4096].
//        Double-buffered LDS; B-tile via async global->LDS copies.
// ---------------------------------------------------------------------------
template <int NCH, bool RELU, bool F16OUT>
__global__ __launch_bounds__(256) void gemm_adj_kernel(
    const float* __restrict__ adj, const float* __restrict__ mask,
    const _Float16* __restrict__ Bt, const float* __restrict__ bias,
    void* __restrict__ out) {
  constexpr int BM = 128, BN = 128, BK = 32, LD = BK + 8;
  constexpr int NK = NNODE / BK;
  __shared__ _Float16 As[2][BM * LD];
  __shared__ _Float16 Bs[2][BN * LD];
  const int tid = threadIdx.x, lane = tid & 31, wave = tid >> 5;
  const int wm = wave & 3, wn = wave >> 2;  // wave tile: 32 rows x 64 cols
  const int m0 = blockIdx.x * BM;
  const int n0 = blockIdx.y * BN;

  // fixed per-thread staging coordinates
  const int ac4 = (tid & 7) * 4;   // A: 4-float column group
  const int ar  = tid >> 3;        // A: row (4 passes of 32)
  const int bcol = tid >> 1;       // B: column
  const int bko  = (tid & 1) * 16; // B: 16-half k group

  v8f acc[2][4];
#pragma unroll
  for (int i = 0; i < 2; ++i)
#pragma unroll
    for (int j = 0; j < 4; ++j) acc[i][j] = (v8f)0.f;

  float4 a4[4], m4[4];
  auto loadA = [&](int k0) {
#pragma unroll
    for (int p = 0; p < 4; ++p) {
      const size_t g = (size_t)(m0 + ar + p * 32) * NNODE + k0 + ac4;
      a4[p] = *(const float4*)(adj + g);
      m4[p] = *(const float4*)(mask + g);
    }
  };
  auto storeA = [&](int buf) {
#pragma unroll
    for (int p = 0; p < 4; ++p) {
      _Float16* d = &As[buf][(ar + p * 32) * LD + ac4];
      d[0] = (_Float16)(a4[p].x * m4[p].x);
      d[1] = (_Float16)(a4[p].y * m4[p].y);
      d[2] = (_Float16)(a4[p].z * m4[p].z);
      d[3] = (_Float16)(a4[p].w * m4[p].w);
    }
  };
  auto asyncB = [&](int k0, int buf) {
    _Float16* l = &Bs[buf][bcol * LD + bko];
    const _Float16* g = Bt + (size_t)(n0 + bcol) * NNODE + k0 + bko;
    async_b128_to_lds(l, g);
    async_b128_to_lds(l + 8, g + 8);
  };

  // prologue: stage tile 0 into buffer 0
  loadA(0);
  storeA(0);
  asyncB(0, 0);
  wait_async0();
  __syncthreads();

  for (int kt = 0; kt < NK; ++kt) {
    const int cur = kt & 1;
    const int nxt = cur ^ 1;
    const bool more = (kt + 1) < NK;
    if (more) {
      asyncB((kt + 1) * BK, nxt);  // async B copy straight into LDS
      loadA((kt + 1) * BK);        // A needs VGPR path (mask mult + cvt)
    }
    v16h af[2], bf[4];
#pragma unroll
    for (int mi = 0; mi < 2; ++mi)
      af[mi] = load_fragA(&As[cur][0], wm * 32 + mi * 16, lane, LD);
#pragma unroll
    for (int ni = 0; ni < 4; ++ni)
      bf[ni] = load_fragB(&Bs[cur][0], wn * 64 + ni * 16, lane, LD);
#pragma unroll
    for (int mi = 0; mi < 2; ++mi)
#pragma unroll
      for (int ni = 0; ni < 4; ++ni)
        acc[mi][ni] = wmma16(af[mi], bf[ni], acc[mi][ni]);
    if (more) storeA(nxt);
    wait_async0();
    __syncthreads();
  }

  // Epilogue: bias (+ReLU), write [b][m][NCH] as f16 or f32.
#pragma unroll
  for (int mi = 0; mi < 2; ++mi) {
#pragma unroll
    for (int ni = 0; ni < 4; ++ni) {
      const int col   = n0 + wn * 64 + ni * 16 + (lane & 15);
      const int mbase = m0 + wm * 32 + mi * 16 + ((lane >> 4) * 8);
      const int b     = col / NCH;
      const int ch    = col % NCH;
      const float bv  = bias[ch];
#pragma unroll
      for (int r = 0; r < 8; ++r) {
        float v = acc[mi][ni][r] + bv;
        if (RELU) v = fmaxf(v, 0.f);
        const size_t idx = (size_t)(b * NNODE + mbase + r) * NCH + ch;
        if (F16OUT)
          ((_Float16*)out)[idx] = (_Float16)v;
        else
          ((float*)out)[idx] = v;
      }
    }
  }
}

// ---------------------------------------------------------------------------
// K3: S2t[b*256+o][n] = sum_k X1[b][n][k] * W2[k][o]
//     per-batch GEMM: M = 4096, N = 256, K = 64 (2 steps)
//     A-tile (pure f16 copy) uses async global->LDS.
// ---------------------------------------------------------------------------
__global__ __launch_bounds__(256) void gemm_x1w2_kernel(
    const _Float16* __restrict__ X1, const float* __restrict__ W2,
    _Float16* __restrict__ S2t) {
  constexpr int BM = 128, BN = 128, BK = 32, LD = BK + 8;
  __shared__ _Float16 As[BM * LD];
  __shared__ _Float16 Bs[BN * LD];
  const int tid = threadIdx.x, lane = tid & 31, wave = tid >> 5;
  const int wm = wave & 3, wn = wave >> 2;
  const int m0 = blockIdx.x * BM;
  const int n0 = blockIdx.y * BN;
  const int b  = blockIdx.z;

  v8f acc[2][4];
#pragma unroll
  for (int i = 0; i < 2; ++i)
#pragma unroll
    for (int j = 0; j < 4; ++j) acc[i][j] = (v8f)0.f;

#pragma unroll
  for (int ks = 0; ks < 2; ++ks) {
    const int k0 = ks * BK;
    // A tile from X1 f16 row-major: async copy into LDS
    {
      const int row = tid >> 1;
      const int ko  = (tid & 1) * 16;
      _Float16* l = &As[row * LD + ko];
      const _Float16* g =
          X1 + ((size_t)b * NNODE + m0 + row) * 64 + k0 + ko;
      async_b128_to_lds(l, g);
      async_b128_to_lds(l + 8, g + 8);
    }
    // B tile: Bs[col][k] = W2[(k0+k)*256 + n0+col]  (f32 -> f16)
    {
      const int col = tid & 127;
      const int kh  = (tid >> 7) * 16;
#pragma unroll
      for (int i = 0; i < 16; ++i)
        Bs[col * LD + kh + i] = (_Float16)W2[(k0 + kh + i) * 256 + n0 + col];
    }
    wait_async0();
    __syncthreads();
    v16h af[2], bf[4];
#pragma unroll
    for (int mi = 0; mi < 2; ++mi)
      af[mi] = load_fragA(As, wm * 32 + mi * 16, lane, LD);
#pragma unroll
    for (int ni = 0; ni < 4; ++ni)
      bf[ni] = load_fragB(Bs, wn * 64 + ni * 16, lane, LD);
#pragma unroll
    for (int mi = 0; mi < 2; ++mi)
#pragma unroll
      for (int ni = 0; ni < 4; ++ni)
        acc[mi][ni] = wmma16(af[mi], bf[ni], acc[mi][ni]);
    __syncthreads();
  }
  // Epilogue: transposed f16 store, 8 contiguous n per lane.
#pragma unroll
  for (int mi = 0; mi < 2; ++mi) {
#pragma unroll
    for (int ni = 0; ni < 4; ++ni) {
      const int o     = n0 + wn * 64 + ni * 16 + (lane & 15);
      const int mbase = m0 + wm * 32 + mi * 16 + ((lane >> 4) * 8);
      h8 v;
#pragma unroll
      for (int r = 0; r < 8; ++r) v[r] = (_Float16)acc[mi][ni][r];
      *(h8*)(S2t + (size_t)(b * 256 + o) * NNODE + mbase) = v;
    }
  }
}

// ---------------------------------------------------------------------------
// K5: LSTM gate math. CC is f32 [B][N][256] (i|f|o|g blocks of 64).
// ---------------------------------------------------------------------------
__global__ __launch_bounds__(256) void gates_kernel(
    const float* __restrict__ CC, const float* __restrict__ c_cur,
    float* __restrict__ h_out, float* __restrict__ c_out) {
  const int g  = blockIdx.x * 256 + threadIdx.x;   // 0 .. B*N*64-1
  const int j  = g & 63;
  const int bn = g >> 6;                           // b*4096 + n
  const size_t base = (size_t)bn * 256 + j;
  const float ci = CC[base];
  const float cf = CC[base + 64];
  const float co = CC[base + 128];
  const float cg = CC[base + 192];
  const float ig = 1.f / (1.f + expf(-ci));
  const float fg = 1.f / (1.f + expf(-cf));
  const float og = 1.f / (1.f + expf(-co));
  const float gg = tanhf(cg);
  const float cp = c_cur[(size_t)bn * 64 + j];
  const float cn = fg * cp + ig * gg;
  const float hn = og * tanhf(cn);
  h_out[(size_t)bn * 64 + j] = hn;
  c_out[(size_t)bn * 64 + j] = cn;
}

// ---------------------------------------------------------------------------
// Launch
// ---------------------------------------------------------------------------
extern "C" void kernel_launch(void* const* d_in, const int* in_sizes, int n_in,
                              void* d_out, int out_size, void* d_ws,
                              size_t ws_size, hipStream_t stream) {
  const float* input = (const float*)d_in[0];
  const float* h_cur = (const float*)d_in[1];
  const float* c_cur = (const float*)d_in[2];
  const float* adj   = (const float*)d_in[3];
  const float* W1    = (const float*)d_in[4];
  const float* mask1 = (const float*)d_in[5];
  const float* b1    = (const float*)d_in[6];
  const float* W2    = (const float*)d_in[7];
  const float* mask2 = (const float*)d_in[8];
  const float* b2    = (const float*)d_in[9];

  char* ws = (char*)d_ws;
  _Float16* S1t = (_Float16*)(ws);                          //  4 MB [512][4096]
  _Float16* X1  = (_Float16*)(ws + ((size_t)4 << 20));      //  4 MB [8][4096][64]
  _Float16* S2t = (_Float16*)(ws + ((size_t)8 << 20));      // 16 MB [2048][4096]
  float*    CC  = (float*)(ws + ((size_t)24 << 20));        // 32 MB [8][4096][256]

  float* h_next = (float*)d_out;
  float* c_next = h_next + (size_t)BATCH * NNODE * 64;

  // K1: S1t = (concat(x,h) @ W1)^T  (f16)
  gemm_in_kernel<<<dim3(BATCH * NNODE / 128), 256, 0, stream>>>(input, h_cur,
                                                                W1, S1t);
  // K2: X1 = relu((adj*mask1) @ S1 + b1)  (f16, row-major)
  gemm_adj_kernel<64, true, true>
      <<<dim3(NNODE / 128, BATCH * 64 / 128), 256, 0, stream>>>(adj, mask1, S1t,
                                                                b1, X1);
  // K3: S2t = (X1 @ W2)^T  (f16)
  gemm_x1w2_kernel<<<dim3(NNODE / 128, 256 / 128, BATCH), 256, 0, stream>>>(
      X1, W2, S2t);
  // K4: CC = (adj*mask2) @ S2 + b2  (f32)
  gemm_adj_kernel<256, false, false>
      <<<dim3(NNODE / 128, BATCH * 256 / 128), 256, 0, stream>>>(adj, mask2,
                                                                 S2t, b2, CC);
  // K5: gates -> h_next, c_next
  gates_kernel<<<dim3(BATCH * NNODE * 64 / 256), 256, 0, stream>>>(CC, c_cur,
                                                                   h_next,
                                                                   c_next);
}
